// QuantBasicBlock_87016037417506
// MI455X (gfx1250) — matrix operand order repaired and checked by reading, not compiled, not run on hardware
//
#include <hip/hip_runtime.h>
#include <hip/hip_bf16.h>
#include <math.h>

// ---------------------------------------------------------------------------
// Types for CDNA5 WMMA
// ---------------------------------------------------------------------------
typedef _Float16 h16;
typedef __attribute__((ext_vector_type(16))) _Float16 v16h;
typedef __attribute__((ext_vector_type(8)))  float    v8f;

#define LDA 40   // LDS row stride (halfs) for A tile (32 data + 8 pad)
#define LDB 40   // LDS col stride (halfs) for B tile (32 data + 8 pad); keeps 16B align

// ---------------------------------------------------------------------------
// Implicit-GEMM conv (NHWC activations, [K][COUT] fp16 weights).
//   M = Nb*HOUT*WOUT rows, N = COUT, K = KH*KW*CIN (ordered kh,kw,c).
//   Workgroup tile 128(M) x 128(N), 8 wave32 waves, wave tile 32x64
//   -> 8x v_wmma_f32_16x16x32_f16 per wave per K=32 step.
//   Double-buffered LDS: 1 barrier per K step, global fetch overlapped
//   with WMMA. All conv geometry is compile-time -> no div/mod in loop.
//   MODE: 0 = BN + ReLU          (conv1)
//         1 = BN only            (shortcut)
//         2 = BN + add sc + ReLU (conv2)
// ---------------------------------------------------------------------------
template<int KH, int KW, int STRIDE, int PAD, int CIN, int HOUT, int WOUT, int MODE>
__global__ __launch_bounds__(256)
void gemm_conv(const h16* __restrict__ Asrc, const h16* __restrict__ Bsrc,
               float* __restrict__ Dout, const float* __restrict__ scbuf,
               const float* __restrict__ bng, const float* __restrict__ bnb,
               const float* __restrict__ bnm, const float* __restrict__ bnv)
{
    constexpr int COUT   = 256;
    constexpr int HIN    = HOUT * STRIDE;
    constexpr int WIN    = WOUT * STRIDE;
    constexpr int CB     = CIN / 32;
    constexpr int NSTEPS = KH * KW * CB;

    __shared__ h16 As[2][128 * LDA];   // [buf][row][k]
    __shared__ h16 Bs[2][128 * LDB];   // [buf][col][k] (transposed on store)

    const int tid  = threadIdx.x;
    const int lane = tid & 31;          // wave32
    const int wave = tid >> 5;          // 0..7
    const int blockM = blockIdx.x * 128;
    const int blockN = blockIdx.y * 128;
    const int mOff = (wave & 3) * 32;
    const int nOff = (wave >> 2) * 64;

    // --- A cooperative-load coords (2 threads/row, 16 halfs each) ---
    const int arow = tid >> 1;          // 0..127
    const int acol = (tid & 1) * 16;    // 0 or 16
    const int mRow = blockM + arow;
    const int nImg = mRow / (HOUT * WOUT);   // compile-time divisor
    const int rem  = mRow % (HOUT * WOUT);
    const int oh   = rem / WOUT;
    const int ow   = rem % WOUT;
    const int ihBase = oh * STRIDE - PAD;
    const int iwBase = ow * STRIDE - PAD;
    const h16* aBase = Asrc + (size_t)nImg * HIN * WIN * CIN + acol;

    // --- B cooperative-load coords: thread owns 4(K) x 4(N) sub-block ---
    const int colg  = (tid & 31) * 4;   // N offset within tile: 0..124
    const int kquad = (tid >> 5) * 4;   // K offset within step:  0..28
    const h16* bPtr = Bsrc + (size_t)kquad * COUT + blockN + colg;

    // ---- incremental fetch state (no div/mod in the loop) ----
    int f_khi = 0, f_kwi = 0, f_cb = 0;

    auto fetchA = [&](uint4& r0, uint4& r1) {
        const int ih = ihBase + f_khi;
        const int iw = iwBase + f_kwi;
        r0.x = r0.y = r0.z = r0.w = 0u;
        r1.x = r1.y = r1.z = r1.w = 0u;
        if (ih >= 0 && ih < HIN && iw >= 0 && iw < WIN) {
            const uint4* gp = (const uint4*)(aBase + ((size_t)ih * WIN + iw) * CIN + f_cb * 32);
            r0 = gp[0];
            r1 = gp[1];
        }
    };
    auto fetchB = [&](uint2* r) {
#pragma unroll
        for (int rr = 0; rr < 4; ++rr)
            r[rr] = *(const uint2*)(bPtr + (size_t)rr * COUT);
    };
    auto advance = [&]() {
        bPtr += 32 * COUT;
        if (++f_cb == CB) {
            f_cb = 0;
            if (++f_kwi == KW) { f_kwi = 0; ++f_khi; }
        }
    };
    auto stageA = [&](int buf, const uint4& r0, const uint4& r1) {
        uint4* lp = (uint4*)(&As[buf][arow * LDA + acol]);
        lp[0] = r0;
        lp[1] = r1;
    };
    // Transpose 4x4 half sub-block in registers, store 4x ds_store_b64
    auto stageB = [&](int buf, const uint2* r) {
        const h16* h0 = (const h16*)&r[0];
        const h16* h1 = (const h16*)&r[1];
        const h16* h2 = (const h16*)&r[2];
        const h16* h3 = (const h16*)&r[3];
#pragma unroll
        for (int c = 0; c < 4; ++c) {
            h16 tmp[4] = { h0[c], h1[c], h2[c], h3[c] };   // 4 K-values, one col
            *(uint2*)(&Bs[buf][(colg + c) * LDB + kquad]) = *(const uint2*)tmp;
        }
    };

    // ---- accumulators: 2 (M) x 4 (N) fragments per wave ----
    v8f acc[2][4];
#pragma unroll
    for (int i = 0; i < 2; ++i)
#pragma unroll
        for (int j = 0; j < 4; ++j)
#pragma unroll
            for (int r = 0; r < 8; ++r) acc[i][j][r] = 0.0f;

    // ---- prologue: stage step 0 ----
    {
        uint4 a0, a1;
        uint2 b[4];
        fetchA(a0, a1);
        fetchB(b);
        stageA(0, a0, a1);
        stageB(0, b);
        advance();
    }
    __syncthreads();

    // ---- pipelined main loop: 1 barrier per K=32 step ----
    for (int s = 0; s < NSTEPS; ++s) {
        const int  cur  = s & 1;
        const bool more = (s + 1) < NSTEPS;

        uint4 a0, a1;
        uint2 b[4];
        if (more) {
            fetchA(a0, a1);
            fetchB(b);
            // prefetch the step-after-next B rows (global_prefetch_b8)
            __builtin_prefetch(bPtr + 32 * COUT, 0, 0);
            advance();
        }

        // fragment loads (ISA 16-bit A/B lane layouts) from current buffer
        v16h aF[2], bF[4];
#pragma unroll
        for (int i = 0; i < 2; ++i) {
            const h16* p  = &As[cur][(mOff + i * 16 + (lane & 15)) * LDA];
            const int  kb = (lane < 16) ? 0 : 8;
#pragma unroll
            for (int v = 0; v < 8; ++v) {
                const int k = kb + (v >> 2) * 16 + (v & 3) * 2;
                aF[i][2 * v]     = p[k];
                aF[i][2 * v + 1] = p[k + 1];
            }
        }
#pragma unroll
        for (int j = 0; j < 4; ++j) {
            const h16* p = &Bs[cur][(nOff + j * 16 + (lane & 15)) * LDB + ((lane < 16) ? 0 : 8)];
#pragma unroll
            for (int v = 0; v < 16; ++v) bF[j][v] = p[v + (v >> 3) * 8];
        }

        // 8 WMMAs per wave per step (A operand reused across j)
#pragma unroll
        for (int i = 0; i < 2; ++i)
#pragma unroll
            for (int j = 0; j < 4; ++j)
                acc[i][j] = __builtin_amdgcn_wmma_f32_16x16x32_f16(
                    false, aF[i], false, bF[j], (short)0, acc[i][j], false, false);

        if (more) {
            stageA(cur ^ 1, a0, a1);
            stageB(cur ^ 1, b);
        }
        __syncthreads();
    }

    // ---- epilogue: BN (+shortcut) (+ReLU), f32 NHWC store ----
    const float eps = 1e-5f;
#pragma unroll
    for (int j = 0; j < 4; ++j) {
        const int cG = blockN + nOff + j * 16 + (lane & 15);
        const float inv  = bng[cG] * rsqrtf(bnv[cG] + eps);
        const float bias = bnb[cG] - bnm[cG] * inv;
#pragma unroll
        for (int i = 0; i < 2; ++i) {
#pragma unroll
            for (int r = 0; r < 8; ++r) {
                const int mG = blockM + mOff + i * 16 + r + ((lane < 16) ? 0 : 8);
                float val = acc[i][j][r] * inv + bias;
                if (MODE == 2) val += scbuf[(size_t)mG * COUT + cG];
                if (MODE != 1) val = fmaxf(val, 0.0f);
                Dout[(size_t)mG * COUT + cG] = val;
            }
        }
    }
}

// ---------------------------------------------------------------------------
// DSConv weight quant: per (o, 32-ch block, tap) int8 with float scale.
// Writes fp16 B-matrix laid out [K][Cout], K ordered (kh, kw, c).
// ---------------------------------------------------------------------------
__global__ void quant_weight_kernel(const float* __restrict__ w, h16* __restrict__ Bq,
                                    int O, int Cin, int KH, int KW)
{
    const int t    = blockIdx.x * blockDim.x + threadIdx.x;
    const int taps = KH * KW;
    const int bpo  = (Cin / 32) * taps;
    if (t >= O * bpo) return;
    const int o   = t / bpo;
    const int r   = t % bpo;
    const int cb  = r / taps;
    const int tap = r % taps;
    const int khi = tap / KW, kwi = tap % KW;

    float mx = 0.0f;
    for (int jj = 0; jj < 32; ++jj) {
        const int ci = cb * 32 + jj;
        const float v = w[(((size_t)o * Cin + ci) * KH + khi) * KW + kwi];
        mx = fmaxf(mx, fabsf(v));
    }
    const float scale = fmaxf(mx, 1e-12f) / 127.0f;
    for (int jj = 0; jj < 32; ++jj) {
        const int ci = cb * 32 + jj;
        const float v = w[(((size_t)o * Cin + ci) * KH + khi) * KW + kwi];
        const float q = fminf(fmaxf(rintf(v / scale), -128.0f), 127.0f);
        Bq[((size_t)(khi * KW + kwi) * Cin + ci) * O + o] = (h16)(q * scale);
    }
}

// ---------------------------------------------------------------------------
// NCHW f32 -> NHWC fp16
// ---------------------------------------------------------------------------
__global__ void nchw_to_nhwc_f16_kernel(const float* __restrict__ x, h16* __restrict__ y,
                                        int Nb, int C, int H, int W)
{
    const size_t idx = (size_t)blockIdx.x * blockDim.x + threadIdx.x;
    const size_t total = (size_t)Nb * C * H * W;
    if (idx >= total) return;
    size_t t = idx;
    const int w = (int)(t % W); t /= W;
    const int h = (int)(t % H); t /= H;
    const int c = (int)(t % C); t /= C;
    const int n = (int)t;
    y[(((size_t)n * H + h) * W + w) * C + c] = (h16)x[idx];
}

// ---------------------------------------------------------------------------
// BFP activation quant over 32-channel blocks (channels contiguous in NHWC).
// Optionally emits fp16 NHWC (next conv's A) and/or f32 NCHW (final output).
// ---------------------------------------------------------------------------
__global__ void bfp_quant_kernel(const float* __restrict__ in, h16* __restrict__ out16,
                                 float* __restrict__ outNCHW, int M, int C, int H, int W)
{
    const int t = blockIdx.x * blockDim.x + threadIdx.x;
    const int nblk = C / 32;
    if (t >= M * nblk) return;
    const int m  = t / nblk;
    const int cb = t % nblk;
    const float* p = in + (size_t)m * C + cb * 32;

    float mx = 0.0f;
#pragma unroll
    for (int j = 0; j < 32; ++j) mx = fmaxf(mx, fabsf(p[j]));
    const float e = floorf(log2f(fmaxf(mx, 1e-12f)));
    const float scale  = exp2f(e - 6.0f);        // e + 1 - (BIT-1), BIT=8
    const float rscale = exp2f(6.0f - e);

    const int HW = H * W;
    const int n = m / HW, rem = m % HW;
    const int hh = rem / W, ww = rem % W;
#pragma unroll
    for (int j = 0; j < 32; ++j) {
        const float q  = fminf(fmaxf(rintf(p[j] * rscale), -128.0f), 127.0f);
        const float xq = q * scale;
        const int c = cb * 32 + j;
        if (out16)   out16[(size_t)m * C + c] = (h16)xq;
        if (outNCHW) outNCHW[(((size_t)n * C + c) * H + hh) * W + ww] = xq;
    }
}

// ---------------------------------------------------------------------------
// Launcher
// ---------------------------------------------------------------------------
extern "C" void kernel_launch(void* const* d_in, const int* in_sizes, int n_in,
                              void* d_out, int out_size, void* d_ws, size_t ws_size,
                              hipStream_t stream)
{
    const float* x    = (const float*)d_in[0];
    const float* w1   = (const float*)d_in[1];
    const float* w2   = (const float*)d_in[2];
    const float* ws   = (const float*)d_in[3];
    const float* bn1g = (const float*)d_in[4];
    const float* bn1b = (const float*)d_in[5];
    const float* bn1m = (const float*)d_in[6];
    const float* bn1v = (const float*)d_in[7];
    const float* bn2g = (const float*)d_in[8];
    const float* bn2b = (const float*)d_in[9];
    const float* bn2m = (const float*)d_in[10];
    const float* bn2v = (const float*)d_in[11];
    const float* bnsg = (const float*)d_in[12];
    const float* bnsb = (const float*)d_in[13];
    const float* bnsm = (const float*)d_in[14];
    const float* bnsv = (const float*)d_in[15];

    constexpr int Nb = 64, Cin = 128, Cout = 256;
    constexpr int H = 56, W = 56, Ho = 28, Wo = 28;
    constexpr int M  = Nb * Ho * Wo;          // 50176
    constexpr int K1 = Cin * 9;               // 1152
    constexpr int K2 = Cout * 9;              // 2304
    constexpr int Ks = Cin;                   // 128

    // ---- scratch carve-out ----
    char* base = (char*)d_ws;
    size_t off = 0;
    h16* xh   = (h16*)(base + off); off += (size_t)Nb * H * W * Cin * sizeof(h16);
    h16* a1h  = (h16*)(base + off); off += (size_t)M * Cout * sizeof(h16);
    h16* w1q  = (h16*)(base + off); off += (size_t)K1 * Cout * sizeof(h16);
    h16* w2q  = (h16*)(base + off); off += (size_t)K2 * Cout * sizeof(h16);
    h16* wsq  = (h16*)(base + off); off += (size_t)Ks * Cout * sizeof(h16);
    float* y1 = (float*)(base + off); off += (size_t)M * Cout * sizeof(float);
    float* sc = (float*)(base + off); off += (size_t)M * Cout * sizeof(float);
    float* y2 = (float*)(base + off); off += (size_t)M * Cout * sizeof(float);

    // 1) x: NCHW f32 -> NHWC fp16
    {
        const size_t total = (size_t)Nb * Cin * H * W;
        nchw_to_nhwc_f16_kernel<<<(unsigned)((total + 255) / 256), 256, 0, stream>>>(
            x, xh, Nb, Cin, H, W);
    }
    // 2) weight quant -> fp16 [K][Cout]
    quant_weight_kernel<<<(256 * (Cin / 32) * 9 + 255) / 256, 256, 0, stream>>>(
        w1, w1q, Cout, Cin, 3, 3);
    quant_weight_kernel<<<(256 * (Cout / 32) * 9 + 255) / 256, 256, 0, stream>>>(
        w2, w2q, Cout, Cout, 3, 3);
    quant_weight_kernel<<<(256 * (Cin / 32) * 1 + 255) / 256, 256, 0, stream>>>(
        ws, wsq, Cout, Cin, 1, 1);

    dim3 gemmGrid(M / 128, Cout / 128);   // (392, 2)

    // 3) conv1 3x3 s2 p1 + BN1 + ReLU -> y1 (NHWC f32)
    gemm_conv<3, 3, 2, 1, 128, 28, 28, 0><<<gemmGrid, 256, 0, stream>>>(
        xh, w1q, y1, nullptr, bn1g, bn1b, bn1m, bn1v);

    // 4) BFP quant -> act1 fp16 (NHWC), A-matrix for conv2
    bfp_quant_kernel<<<(M * (Cout / 32) + 255) / 256, 256, 0, stream>>>(
        y1, a1h, nullptr, M, Cout, Ho, Wo);

    // 5) shortcut 1x1 s2 + BNs -> sc (NHWC f32)
    gemm_conv<1, 1, 2, 0, 128, 28, 28, 1><<<gemmGrid, 256, 0, stream>>>(
        xh, wsq, sc, nullptr, bnsg, bnsb, bnsm, bnsv);

    // 6) conv2 3x3 s1 p1 + BN2 + add sc + ReLU -> y2 (NHWC f32)
    gemm_conv<3, 3, 1, 1, 256, 28, 28, 2><<<gemmGrid, 256, 0, stream>>>(
        a1h, w2q, y2, sc, bn2g, bn2b, bn2m, bn2v);

    // 7) final BFP quant, NHWC f32 -> NCHW f32 into d_out
    bfp_quant_kernel<<<(M * (Cout / 32) + 255) / 256, 256, 0, stream>>>(
        y2, nullptr, (float*)d_out, M, Cout, Ho, Wo);
}
